// Dec_18373870092832
// MI455X (gfx1250) — matrix operand order repaired and checked by reading, not compiled, not run on hardware
//
#include <hip/hip_runtime.h>
#include <hip/hip_bf16.h>

// ---------------------------------------------------------------------------
// MI455X (gfx1250): hypernetwork + ConvTranspose3d decoder.
// Heavy math on v_wmma_f32_16x16x32_f16 (wave32 WMMA), channels-last f16
// activations, fp32 accumulation. Register-blocked 32x64 (or 32x32) wave
// tiles: 12 b128 loads feed 8 wmma per K-chunk (load:wmma = 1.5).
// ---------------------------------------------------------------------------

typedef __attribute__((ext_vector_type(16))) _Float16 v16h;
typedef __attribute__((ext_vector_type(8)))  float    v8f;

union Frag16 { v16h v; uint4 q[2]; };
union H8 { uint4 q; _Float16 h[8]; };

// A fragment (16x32 f16, row-major M x K source):
// lanes 0-15: M=lane, halves[0..7]=K kb..kb+7, halves[8..15]=K kb+16..kb+23
// lanes16-31: M=lane-16, K offset +8
__device__ __forceinline__ v16h load_a_frag(const _Float16* rowptr, int kbase, int lane) {
    const int kh = kbase + ((lane >> 4) << 3);
    Frag16 f;
    f.q[0] = *(const uint4*)(rowptr + kh);
    f.q[1] = *(const uint4*)(rowptr + kh + 16);
    return f.v;
}

// B fragment (32x16 f16). B stored as [N][K] row-major (cout rows, cin contig):
// lanes 0-15: N=lane, halves[0..7]=K kb..kb+7, halves[8..15]=K kb+16..kb+23 ; lanes 16-31: N=lane-16, K offset +8
__device__ __forceinline__ v16h load_b_frag(const _Float16* colptr, int kbase, int lane) {
    const int kh = kbase + ((lane >> 4) << 3);
    Frag16 f;
    f.q[0] = *(const uint4*)(colptr + kh);
    f.q[1] = *(const uint4*)(colptr + kh + 16);
    return f.v;
}

__device__ __forceinline__ v16h zero_frag() {
    Frag16 f;
    f.q[0] = uint4{0u, 0u, 0u, 0u};
    f.q[1] = uint4{0u, 0u, 0u, 0u};
    return f.v;
}

// ---------------------------------------------------------------------------
// Hypernetwork: wB[b][h][o][c] = sum_q y[b][q] * tables[h][q][o*1024+c]  (f16)
// ---------------------------------------------------------------------------
__global__ __launch_bounds__(256) void hyper_kernel(
    const float* __restrict__ y, const float* __restrict__ tables,
    _Float16* __restrict__ wB) {
    const unsigned i = blockIdx.x * 256u + threadIdx.x;   // 2^21 total
    if (i >= 2097152u) return;
    const int c = i & 1023;
    const int o = (i >> 10) & 127;
    const int h = (i >> 17) & 7;
    const int b = i >> 20;
    const float* tb = tables + ((size_t)h * 24) * 131072 + (size_t)o * 1024 + c;
    const float* yb = y + b * 24;
    float s = 0.f;
#pragma unroll
    for (int q = 0; q < 24; ++q) s += yb[q] * tb[(size_t)q * 131072];
    wB[i] = (_Float16)s;
}

// x_in [B][1024][192] -> xA [B][192][1024] f16 (channels-last A matrix)
__global__ __launch_bounds__(256) void xa_kernel(
    const float* __restrict__ x, _Float16* __restrict__ xA) {
    const unsigned i = blockIdx.x * 256u + threadIdx.x;
    if (i >= 393216u) return;
    const int c = i & 1023;
    const int s = (i >> 10) % 192;
    const int b = (i >> 10) / 192;
    xA[i] = (_Float16)x[((size_t)b * 1024 + c) * 192 + s];
}

// w [Cin][Cout][64] f32 -> wt [tap][Cout][Cin] f16
__global__ __launch_bounds__(256) void wtr_kernel(
    const float* __restrict__ w, _Float16* __restrict__ wt,
    int Cin, int Cout, unsigned total) {
    const unsigned i = blockIdx.x * 256u + threadIdx.x;
    if (i >= total) return;
    const int cin  = i % Cin;
    const unsigned t2 = i / Cin;
    const int cout = t2 % Cout;
    const int tap  = t2 / Cout;
    wt[i] = (_Float16)w[((size_t)cin * Cout + cout) * 64 + tap];
}

// ---------------------------------------------------------------------------
// Projection GEMM (register-blocked 32x64): per (b, head):
// D[192x128] = A[192x1024] * B[128x1024]^T -> act0[b][s][head*128+o] f16
// ---------------------------------------------------------------------------
__global__ __launch_bounds__(256) void proj_wmma(
    const _Float16* __restrict__ xA, const _Float16* __restrict__ wB,
    _Float16* __restrict__ out0) {
    constexpr int MB = 2, NB = 4;
    const int wave = (int)((blockIdx.x * 256u + threadIdx.x) >> 5);
    if (wave >= 2 * 8 * 6 * 2) return;   // B * heads * (192/32) * (128/64)
    const int lane = threadIdx.x & 31;
    int t = wave;
    const int nt = t & 1;  t >>= 1;
    const int mt = t % 6;  t /= 6;
    const int head = t & 7; t >>= 3;
    const int b = t;

    const _Float16* A  = xA + (size_t)b * 192 * 1024;
    const _Float16* Bm = wB + ((size_t)(b * 8 + head) * 128) * 1024;

    const _Float16* arow[MB];
    const _Float16* brow[NB];
    int colc[NB];
#pragma unroll
    for (int i = 0; i < MB; ++i)
        arow[i] = A + (size_t)((mt * MB + i) * 16 + (lane & 15)) * 1024;
#pragma unroll
    for (int j = 0; j < NB; ++j) {
        colc[j] = (nt * NB + j) * 16 + (lane & 15);
        brow[j] = Bm + (size_t)colc[j] * 1024;
    }

    v8f acc[MB][NB];
#pragma unroll
    for (int i = 0; i < MB; ++i)
#pragma unroll
        for (int j = 0; j < NB; ++j)
            acc[i][j] = v8f{0.f, 0.f, 0.f, 0.f, 0.f, 0.f, 0.f, 0.f};

#pragma unroll 2
    for (int cb = 0; cb < 1024; cb += 32) {
        v16h a[MB], bf[NB];
#pragma unroll
        for (int i = 0; i < MB; ++i) a[i] = load_a_frag(arow[i], cb, lane);
#pragma unroll
        for (int j = 0; j < NB; ++j) bf[j] = load_b_frag(brow[j], cb, lane);
#pragma unroll
        for (int i = 0; i < MB; ++i)
#pragma unroll
            for (int j = 0; j < NB; ++j)
                acc[i][j] = __builtin_amdgcn_wmma_f32_16x16x32_f16(
                    false, a[i], false, bf[j], (short)0, acc[i][j], false, false);
    }

#pragma unroll
    for (int i = 0; i < MB; ++i) {
#pragma unroll
        for (int r = 0; r < 8; ++r) {
            const int m = (mt * MB + i) * 16 + r + ((lane >> 4) << 3);
            _Float16* dst = out0 + ((size_t)b * 192 + m) * 1024 + head * 128;
#pragma unroll
            for (int j = 0; j < NB; ++j)
                dst[colc[j]] = (_Float16)acc[i][j][r];
        }
    }
}

// ---------------------------------------------------------------------------
// ConvTranspose3d(k=4,s=2,p=1) as implicit WMMA GEMM over 8 parity classes.
// in:  [B][Di][Hi][Wi][Cin]  f16   wt: [tap64][Cout][Cin] f16
// out: [B][2Di][2Hi][2Wi][Cout] f16 (bias added).  Wave tile: (MB*16)x(NB*16).
// ---------------------------------------------------------------------------
template <int Cin, int Cout, int MB, int NB>
__global__ __launch_bounds__(256) void convt_wmma(
    const _Float16* __restrict__ in, const _Float16* __restrict__ wt,
    const float* __restrict__ bias, _Float16* __restrict__ out,
    int Di, int Hi, int Wi) {
    const int Do = 2 * Di, Ho = 2 * Hi, Wo = 2 * Wi;
    const int Msub   = Di * Hi * Wi;
    const int Mtiles = Msub / (16 * MB);
    const int Ntiles = Cout / (16 * NB);
    const int total  = 2 * 8 * Mtiles * Ntiles;
    const int wave = (int)((blockIdx.x * 256u + threadIdx.x) >> 5);
    if (wave >= total) return;
    const int lane = threadIdx.x & 31;

    int t = wave;
    const int nt = t % Ntiles; t /= Ntiles;
    const int mt = t % Mtiles; t /= Mtiles;
    const int pc = t & 7;      t >>= 3;
    const int b  = t;
    const int pd = (pc >> 2) & 1, ph = (pc >> 1) & 1, pw = pc & 1;

    int ud[MB], uh[MB], uw[MB];
#pragma unroll
    for (int i = 0; i < MB; ++i) {
        const int m = (mt * MB + i) * 16 + (lane & 15);
        uw[i] = m % Wi;
        uh[i] = (m / Wi) % Hi;
        ud[i] = m / (Wi * Hi);
    }
    int coutL[NB];
#pragma unroll
    for (int j = 0; j < NB; ++j) coutL[j] = (nt * NB + j) * 16 + (lane & 15);

    v8f acc[MB][NB];
#pragma unroll
    for (int i = 0; i < MB; ++i)
#pragma unroll
        for (int j = 0; j < NB; ++j)
            acc[i][j] = v8f{0.f, 0.f, 0.f, 0.f, 0.f, 0.f, 0.f, 0.f};

#pragma unroll
    for (int tap = 0; tap < 8; ++tap) {
        const int td = (tap >> 2) & 1, th = (tap >> 1) & 1, tw = tap & 1;
        const int kd = pd ? 2 - 2 * td : 1 + 2 * td;
        const int kh = ph ? 2 - 2 * th : 1 + 2 * th;
        const int kw = pw ? 2 - 2 * tw : 1 + 2 * tw;
        const int ktap = (kd * 4 + kh) * 4 + kw;

        const _Float16* arow[MB];
        bool valid[MB];
#pragma unroll
        for (int i = 0; i < MB; ++i) {
            // o = 2i - 1 + k ; per parity p, tap t: in = u + (t? (p? +1 : -1) : 0)
            const int id = ud[i] + (td ? (pd ? 1 : -1) : 0);
            const int ih = uh[i] + (th ? (ph ? 1 : -1) : 0);
            const int iw = uw[i] + (tw ? (pw ? 1 : -1) : 0);
            valid[i] = (unsigned)id < (unsigned)Di &&
                       (unsigned)ih < (unsigned)Hi &&
                       (unsigned)iw < (unsigned)Wi;
            arow[i] = in + (size_t)(((b * Di + id) * Hi + ih) * Wi + iw) * Cin;
        }
        const _Float16* brow[NB];
#pragma unroll
        for (int j = 0; j < NB; ++j)
            brow[j] = wt + ((size_t)ktap * Cout + coutL[j]) * Cin;

#pragma unroll 2
        for (int cb = 0; cb < Cin; cb += 32) {
            __builtin_prefetch(brow[0] + cb + 128, 0, 1);   // global_prefetch_b8
            v16h a[MB], bf[NB];
#pragma unroll
            for (int i = 0; i < MB; ++i)
                a[i] = valid[i] ? load_a_frag(arow[i], cb, lane) : zero_frag();
#pragma unroll
            for (int j = 0; j < NB; ++j) bf[j] = load_b_frag(brow[j], cb, lane);
#pragma unroll
            for (int i = 0; i < MB; ++i)
#pragma unroll
                for (int j = 0; j < NB; ++j)
                    acc[i][j] = __builtin_amdgcn_wmma_f32_16x16x32_f16(
                        false, a[i], false, bf[j], (short)0, acc[i][j], false, false);
        }
    }

    float bv[NB];
#pragma unroll
    for (int j = 0; j < NB; ++j) bv[j] = bias[coutL[j]];
#pragma unroll
    for (int i = 0; i < MB; ++i) {
#pragma unroll
        for (int r = 0; r < 8; ++r) {
            const int mm  = (mt * MB + i) * 16 + r + ((lane >> 4) << 3);
            const int uw2 = mm % Wi;
            const int rem = mm / Wi;
            const int uh2 = rem % Hi;
            const int ud2 = rem / Hi;
            const int od = 2 * ud2 + pd, oh = 2 * uh2 + ph, ow = 2 * uw2 + pw;
            _Float16* dst = out + (size_t)(((b * Do + od) * Ho + oh) * Wo + ow) * Cout;
#pragma unroll
            for (int j = 0; j < NB; ++j)
                dst[coutL[j]] = (_Float16)(acc[i][j][r] + bv[j]);
        }
    }
}

// ---------------------------------------------------------------------------
// BatchNorm (training mode, biased var) over channels-last [S][C] f16.
// ---------------------------------------------------------------------------
__global__ __launch_bounds__(256) void zero_kernel(float* p, int n) {
    const int i = blockIdx.x * 256 + threadIdx.x;
    if (i < n) p[i] = 0.f;
}

template <int C>
__global__ __launch_bounds__(256) void stats_kernel(
    const _Float16* __restrict__ act, float* __restrict__ stats,
    int S, int rowsPerBlock) {
    constexpr int Cw     = (C < 256) ? C : 256;
    constexpr int CPT    = (C + 255) / 256;
    constexpr int groups = 256 / Cw;
    __shared__ float red[2][256];
    const int t  = threadIdx.x;
    const int cl = t % Cw;
    const int g  = t / Cw;
    const int r0 = blockIdx.x * rowsPerBlock;
    const int r1 = (r0 + rowsPerBlock < S) ? (r0 + rowsPerBlock) : S;
    float sum[CPT] = {};
    float sq[CPT]  = {};
    for (int r = r0 + g; r < r1; r += groups) {
        const _Float16* row = act + (size_t)r * C;
#pragma unroll
        for (int j = 0; j < CPT; ++j) {
            const float v = (float)row[cl + j * 256];
            sum[j] += v;
            sq[j]  += v * v;
        }
    }
    if (groups > 1) {
        red[0][t] = sum[0];
        red[1][t] = sq[0];
        __syncthreads();
        if (g == 0) {
            float a = sum[0], b2 = sq[0];
            for (int gg = 1; gg < groups; ++gg) {
                a  += red[0][cl + gg * Cw];
                b2 += red[1][cl + gg * Cw];
            }
            atomicAdd(&stats[cl], a);
            atomicAdd(&stats[C + cl], b2);
        }
    } else {
#pragma unroll
        for (int j = 0; j < CPT; ++j) {
            atomicAdd(&stats[cl + j * 256], sum[j]);
            atomicAdd(&stats[C + cl + j * 256], sq[j]);
        }
    }
}

__global__ __launch_bounds__(256) void bnfin_kernel(
    float* __restrict__ stats, const float* __restrict__ g,
    const float* __restrict__ be, int C, float invN) {
    const int c = blockIdx.x * 256 + threadIdx.x;
    if (c >= C) return;
    const float mean = stats[c] * invN;
    const float var  = stats[C + c] * invN - mean * mean;
    const float rstd = rsqrtf(var + 1e-5f);
    const float sc = g[c] * rstd;
    stats[c]     = sc;
    stats[C + c] = be[c] - mean * sc;   // y = x*sc + shift
}

// In-place normalize+relu, 8 halves (one b128) per thread.
__global__ __launch_bounds__(256) void bnrelu_kernel(
    _Float16* __restrict__ act, const float* __restrict__ stats,
    int C, size_t total8) {
    const size_t i = (size_t)blockIdx.x * 256 + threadIdx.x;
    if (i >= total8) return;
    const size_t base = i * 8;
    const int c0 = (int)(base % C);
    H8 v;
    v.q = *(const uint4*)(act + base);
#pragma unroll
    for (int k = 0; k < 8; ++k) {
        const float x = (float)v.h[k] * stats[c0 + k] + stats[C + c0 + k];
        v.h[k] = (_Float16)(x > 0.f ? x : 0.f);
    }
    *(uint4*)(act + base) = v.q;
}

// ---------------------------------------------------------------------------
// Final ConvT (32 -> 1) + sigmoid.  Cout=1 -> VALU dot with weights in LDS.
// ---------------------------------------------------------------------------
__global__ __launch_bounds__(256) void convt5_kernel(
    const _Float16* __restrict__ act, const float* __restrict__ w5,
    const float* __restrict__ b5, float* __restrict__ out) {
    __shared__ float ws5[64 * 32];
    for (int i = threadIdx.x; i < 2048; i += 256) {
        const int tap = i >> 5, c = i & 31;
        ws5[i] = w5[c * 64 + tap];   // w5 [32][1][64]
    }
    __syncthreads();
    const size_t idx = (size_t)blockIdx.x * 256 + threadIdx.x;
    if (idx >= 12582912ull) return;
    const int ow = (int)(idx % 256);
    size_t t2 = idx / 256;
    const int oh = (int)(t2 % 192); t2 /= 192;
    const int od = (int)(t2 % 128);
    const int b  = (int)(t2 / 128);
    const int pd = od & 1, ph = oh & 1, pw = ow & 1;
    const int ud = od >> 1, uh = oh >> 1, uw = ow >> 1;
    const int Di = 64, Hi = 96, Wi = 128;
    float accum = b5[0];
#pragma unroll
    for (int tap = 0; tap < 8; ++tap) {
        const int td = (tap >> 2) & 1, th = (tap >> 1) & 1, tw = tap & 1;
        const int id = ud + (td ? (pd ? 1 : -1) : 0);
        const int ih = uh + (th ? (ph ? 1 : -1) : 0);
        const int iw = uw + (tw ? (pw ? 1 : -1) : 0);
        if ((unsigned)id >= (unsigned)Di || (unsigned)ih >= (unsigned)Hi ||
            (unsigned)iw >= (unsigned)Wi) continue;
        const int kd = pd ? 2 - 2 * td : 1 + 2 * td;
        const int kh = ph ? 2 - 2 * th : 1 + 2 * th;
        const int kw = pw ? 2 - 2 * tw : 1 + 2 * tw;
        const float* wk = &ws5[((kd * 4 + kh) * 4 + kw) * 32];
        const _Float16* row = act + (size_t)(((b * Di + id) * Hi + ih) * Wi + iw) * 32;
#pragma unroll
        for (int c = 0; c < 32; ++c) accum += (float)row[c] * wk[c];
    }
    out[idx] = 1.f / (1.f + __expf(-accum));
}

// ---------------------------------------------------------------------------
// Host side
// ---------------------------------------------------------------------------
static void runBN(hipStream_t stream, _Float16* act, float* stats, int C, int S,
                  const float* g, const float* be) {
    zero_kernel<<<(2 * C + 255) / 256, 256, 0, stream>>>(stats, 2 * C);
    const int rpb = 2048;
    const int nb = (S + rpb - 1) / rpb;
    switch (C) {
        case 1024: stats_kernel<1024><<<nb, 256, 0, stream>>>(act, stats, S, rpb); break;
        case 512:  stats_kernel<512><<<nb, 256, 0, stream>>>(act, stats, S, rpb); break;
        case 256:  stats_kernel<256><<<nb, 256, 0, stream>>>(act, stats, S, rpb); break;
        case 128:  stats_kernel<128><<<nb, 256, 0, stream>>>(act, stats, S, rpb); break;
        case 32:   stats_kernel<32><<<nb, 256, 0, stream>>>(act, stats, S, rpb); break;
    }
    bnfin_kernel<<<(C + 255) / 256, 256, 0, stream>>>(stats, g, be, C, 1.f / (float)S);
    const size_t total8 = (size_t)S * C / 8;
    bnrelu_kernel<<<(unsigned)((total8 + 255) / 256), 256, 0, stream>>>(act, stats, C, total8);
}

extern "C" void kernel_launch(void* const* d_in, const int* in_sizes, int n_in,
                              void* d_out, int out_size, void* d_ws, size_t ws_size,
                              hipStream_t stream) {
    (void)in_sizes; (void)n_in; (void)out_size; (void)ws_size;
    const float* x_in   = (const float*)d_in[0];
    const float* y_in   = (const float*)d_in[1];
    const float* tables = (const float*)d_in[2];
    const float* g0  = (const float*)d_in[3];
    const float* be0 = (const float*)d_in[4];
    const float* g1  = (const float*)d_in[5];
    const float* be1 = (const float*)d_in[6];
    const float* g2  = (const float*)d_in[7];
    const float* be2 = (const float*)d_in[8];
    const float* g3  = (const float*)d_in[9];
    const float* be3 = (const float*)d_in[10];
    const float* g4  = (const float*)d_in[11];
    const float* be4 = (const float*)d_in[12];
    const float* w1 = (const float*)d_in[13];
    const float* b1 = (const float*)d_in[14];
    const float* w2 = (const float*)d_in[15];
    const float* b2 = (const float*)d_in[16];
    const float* w3 = (const float*)d_in[17];
    const float* b3 = (const float*)d_in[18];
    const float* w4 = (const float*)d_in[19];
    const float* b4 = (const float*)d_in[20];
    const float* w5 = (const float*)d_in[21];
    const float* b5 = (const float*)d_in[22];

    char* ws = (char*)d_ws;
    constexpr size_t OFF_XA    = 0;                          // 786432
    constexpr size_t OFF_WB    = OFF_XA    + 786432;         // 4194304
    constexpr size_t OFF_ACT0  = OFF_WB    + 4194304;        // 786432
    constexpr size_t OFF_ACT1  = OFF_ACT0  + 786432;         // 3145728
    constexpr size_t OFF_ACT2  = OFF_ACT1  + 3145728;        // 12582912
    constexpr size_t OFF_ACT3  = OFF_ACT2  + 12582912;       // 50331648
    constexpr size_t OFF_ACT4  = OFF_ACT3  + 50331648;       // 100663296
    constexpr size_t OFF_WT1   = OFF_ACT4  + 100663296;      // 67108864
    constexpr size_t OFF_WT2   = OFF_WT1   + 67108864;       // 16777216
    constexpr size_t OFF_WT3   = OFF_WT2   + 16777216;       // 4194304
    constexpr size_t OFF_WT4   = OFF_WT3   + 4194304;        // 524288
    constexpr size_t OFF_STATS = OFF_WT4   + 524288;         // 16384

    _Float16* xA   = (_Float16*)(ws + OFF_XA);
    _Float16* wB   = (_Float16*)(ws + OFF_WB);
    _Float16* act0 = (_Float16*)(ws + OFF_ACT0);
    _Float16* act1 = (_Float16*)(ws + OFF_ACT1);
    _Float16* act2 = (_Float16*)(ws + OFF_ACT2);
    _Float16* act3 = (_Float16*)(ws + OFF_ACT3);
    _Float16* act4 = (_Float16*)(ws + OFF_ACT4);
    _Float16* wt1  = (_Float16*)(ws + OFF_WT1);
    _Float16* wt2  = (_Float16*)(ws + OFF_WT2);
    _Float16* wt3  = (_Float16*)(ws + OFF_WT3);
    _Float16* wt4  = (_Float16*)(ws + OFF_WT4);
    float*    stats = (float*)(ws + OFF_STATS);
    float*    out  = (float*)d_out;

    // Weight repack + hypernetwork + input transpose
    wtr_kernel<<<(33554432u + 255) / 256, 256, 0, stream>>>(w1, wt1, 1024, 512, 33554432u);
    wtr_kernel<<<(8388608u  + 255) / 256, 256, 0, stream>>>(w2, wt2, 512, 256, 8388608u);
    wtr_kernel<<<(2097152u  + 255) / 256, 256, 0, stream>>>(w3, wt3, 256, 128, 2097152u);
    wtr_kernel<<<(262144u   + 255) / 256, 256, 0, stream>>>(w4, wt4, 128, 32, 262144u);
    hyper_kernel<<<2097152u / 256, 256, 0, stream>>>(y_in, tables, wB);
    xa_kernel<<<393216u / 256, 256, 0, stream>>>(x_in, xA);

    // Projection (WMMA, 32x64 wave tiles) -> act0 [2][192][1024]
    proj_wmma<<<24, 256, 0, stream>>>(xA, wB, act0);   // 192 waves
    runBN(stream, act0, stats, 1024, 2 * 192, g0, be0);

    // Decoder stack (WMMA implicit GEMMs)
    convt_wmma<1024, 512, 2, 4><<<96,   256, 0, stream>>>(act0, wt1, b1, act1, 4, 6, 8);
    runBN(stream, act1, stats, 512, 2 * 8 * 12 * 16, g1, be1);

    convt_wmma<512, 256, 2, 4><<<384,  256, 0, stream>>>(act1, wt2, b2, act2, 8, 12, 16);
    runBN(stream, act2, stats, 256, 2 * 16 * 24 * 32, g2, be2);

    convt_wmma<256, 128, 2, 4><<<1536, 256, 0, stream>>>(act2, wt3, b3, act3, 16, 24, 32);
    runBN(stream, act3, stats, 128, 2 * 32 * 48 * 64, g3, be3);

    convt_wmma<128, 32, 2, 2><<<6144, 256, 0, stream>>>(act3, wt4, b4, act4, 32, 48, 64);
    runBN(stream, act4, stats, 32, 2 * 64 * 96 * 128, g4, be4);

    // Final layer + sigmoid -> d_out [2][1][128][192][256]
    convt5_kernel<<<49152, 256, 0, stream>>>(act4, w5, b5, out);
}
